// SENNGC_78323023610434
// MI455X (gfx1250) — matrix operand order repaired and checked by reading, not compiled, not run on hardware
//
#include <hip/hip_runtime.h>
#include <hip/hip_fp16.h>
#include <math.h>

// ---------------------------------------------------------------------------
// SENNGC forward for MI455X (gfx1250, wave32, WMMA).
// f16 WMMA (V_WMMA_F32_16X16X32_F16) with f32 accumulation for all GEMMs.
// ---------------------------------------------------------------------------

typedef __attribute__((ext_vector_type(16))) _Float16 v16h;
typedef __attribute__((ext_vector_type(8)))  _Float16 v8h;
typedef __attribute__((ext_vector_type(8)))  float    v8f;

constexpr int BSZ   = 4096;
constexpr int KORD  = 5;      // ORDER
constexpr int NVAR  = 32;     // NV
constexpr int DDIM  = 16;     // DD
constexpr int HID   = 256;    // H
constexpr int INDIM = 176;    // NV*ORDER + DD
constexpr int KP0   = 192;    // INDIM padded to multiple of 32
constexpr int NOUT  = 2048;   // NV*NV*2
constexpr float TEMP_INV = 20.0f;   // 1/0.05
constexpr float GEPS     = 1e-20f;
constexpr float BN_EPS   = 1e-5f;

// --------------------------- fragment loaders ------------------------------
// A-fragment (16x32 f16, wave32): lane L holds row M=(L&15).
// elems 0..7  = K kt*32 + (L>>4)*8 + 0..7
// elems 8..15 = K kt*32 + 16 + (L>>4)*8 + 0..7        (ISA 7.12.2)
__device__ inline v16h load_a_frag(const _Float16* __restrict__ A, int ldk,
                                   int mBase, int kt, int lane) {
  const int row  = mBase + (lane & 15);
  const int half = lane >> 4;
  const _Float16* p = A + (size_t)row * ldk + kt * 32 + half * 8;
  v8h lo = *(const v8h*)(p);
  v8h hi = *(const v8h*)(p + 16);
  v16h r;
#pragma unroll
  for (int e = 0; e < 8; ++e) { r[e] = lo[e]; r[8 + e] = hi[e]; }
  return r;
}

// ----------------------------- weight packing ------------------------------
// Pack W[k][Kreal x N] (f32, row-major) into per-fragment f16 B layout:
// frag (ntile, kt): lane L holds column n = ntile*16 + (L&15),
// elems 0..7 = K kt*32 + (L>>4)*8 + 0..7, elems 8..15 = K kt*32 + 16 + (L>>4)*8 + 0..7.
__global__ void pack_w_kernel(const float* __restrict__ W, _Float16* __restrict__ out,
                              int Kreal, int Kpad, int N) {
  const int tid = blockIdx.x * blockDim.x + threadIdx.x;
  if (tid >= Kpad * N) return;
  const int e     = tid & 15;
  const int lane  = (tid >> 4) & 31;
  const int rem   = tid >> 9;
  const int nkt   = Kpad / 32;
  const int kt    = rem % nkt;
  const int ntile = rem / nkt;
  const int n = ntile * 16 + (lane & 15);
  const int k = kt * 32 + (lane >> 4) * 8 + (e & 7) + (e >> 3) * 16;
  const float v = (k < Kreal) ? W[(size_t)k * N + n] : 0.0f;
  out[tid] = (_Float16)v;
}

// ------------------------------- build x -----------------------------------
// x[b, 0:160]  : lag i<k -> struct_i[b] @ inputs[b,i,:], else passthrough
// x[b,160:176] : dom_var[k]; x[b,176:192] : zero pad. Output f16 row-major.
__global__ void build_x_kernel(const float* __restrict__ inputs,
                               const float* __restrict__ coeffs,
                               const float* __restrict__ src_var,
                               const float* __restrict__ tgt_var,
                               const int*   __restrict__ is_source,
                               _Float16* __restrict__ x, int k) {
  const int b = blockIdx.x;
  const int j = threadIdx.x;              // 0..191
  float val = 0.0f;
  if (j < KORD * NVAR) {
    const int i = j >> 5, v = j & 31;
    if (i < k) {
      const float* st = coeffs + ((size_t)b * KORD + i) * (NVAR * NVAR) + v * NVAR;
      const float* in = inputs + ((size_t)b * KORD + i) * NVAR;
      float s = 0.0f;
#pragma unroll
      for (int jj = 0; jj < NVAR; ++jj) s += st[jj] * in[jj];
      val = s;
    } else {
      val = inputs[((size_t)b * KORD + i) * NVAR + v];
    }
  } else if (j < INDIM) {
    const float* dv = (*is_source) ? src_var : tgt_var;
    val = dv[k * DDIM + (j - (INDIM - DDIM))];
  }
  x[(size_t)b * KP0 + j] = (_Float16)val;
}

// ------------------------------ WMMA GEMM ----------------------------------
// C[B,N] = A[B,KPAD]*Bp + bias. 8 waves/block, each owns one 16x16 N-tile;
// block tile = 16 rows x 128 cols. K loop fully unrolled (KPAD is template).
template <int KPAD>
__global__ void gemm_wmma_kernel(const _Float16* __restrict__ A,
                                 const _Float16* __restrict__ Bp,
                                 const float* __restrict__ bias,
                                 float*    __restrict__ Cf32,
                                 _Float16* __restrict__ Cf16, int N) {
  constexpr int NKT = KPAD / 32;
  const int lane  = threadIdx.x & 31;
  const int wave  = threadIdx.x >> 5;
  const int mBase = blockIdx.x * 16;
  const int ntile = blockIdx.y * 8 + wave;
  v8f acc = {};
  const _Float16* bp = Bp + (size_t)ntile * NKT * 512 + lane * 16;
#pragma unroll
  for (int kt = 0; kt < NKT; ++kt) {
    v16h a = load_a_frag(A, KPAD, mBase, kt, lane);
    v16h b = *(const v16h*)(bp + kt * 512);
    acc = __builtin_amdgcn_wmma_f32_16x16x32_f16(false, a, false, b,
                                                 (short)0, acc, false, false);
  }
  const int n    = ntile * 16 + (lane & 15);
  const int mOff = (lane >> 4) * 8;       // C layout: lanes16-31 hold M=8..15
  const float bv = bias[n];
#pragma unroll
  for (int r = 0; r < 8; ++r) {
    const int m = mBase + mOff + r;
    const float val = acc[r] + bv;
    if (Cf32) Cf32[(size_t)m * N + n] = val;
    if (Cf16) Cf16[(size_t)m * N + n] = (_Float16)val;
  }
}

// ------------------- fused GEMM2 + Gumbel softmax (hard ST) ----------------
// logits[B,2048] tile in f32 accumulators; 2-way softmax pairs are adjacent
// columns == adjacent lanes of the C-fragment -> __shfl_xor(z,1).
__global__ void gemm2_gumbel_kernel(const _Float16* __restrict__ A,
                                    const _Float16* __restrict__ Bp,
                                    const float* __restrict__ bias,
                                    const float* __restrict__ gu,   // + k*B*2048
                                    const int*   __restrict__ hardp,
                                    float* __restrict__ coeffs,
                                    float* __restrict__ all_lag, int k) {
  constexpr int NKT = HID / 32;
  const int lane  = threadIdx.x & 31;
  const int wave  = threadIdx.x >> 5;
  const int mBase = blockIdx.x * 16;
  const int ntile = blockIdx.y * 8 + wave;
  v8f acc = {};
  const _Float16* bp = Bp + (size_t)ntile * NKT * 512 + lane * 16;
#pragma unroll
  for (int kt = 0; kt < NKT; ++kt) {
    v16h a = load_a_frag(A, HID, mBase, kt, lane);
    v16h b = *(const v16h*)(bp + kt * 512);
    acc = __builtin_amdgcn_wmma_f32_16x16x32_f16(false, a, false, b,
                                                 (short)0, acc, false, false);
  }
  const int cc   = ntile * 16 + (lane & 15);  // flat col in [0,2048) == i*64+j*2+s
  const int mOff = (lane >> 4) * 8;
  const float bv = bias[cc];
  const int  hard = *hardp;
  const int  s    = cc & 1;
#pragma unroll
  for (int r = 0; r < 8; ++r) {
    const int   b     = mBase + mOff + r;
    const float logit = acc[r] + bv;
    const float u     = gu[(size_t)b * NOUT + cc];
    const float g     = -__logf(-__logf(u + GEPS) + GEPS);
    const float z     = (logit + g) * TEMP_INV;
    const float zp    = __shfl_xor(z, 1, 32);
    float y = 1.0f / (1.0f + __expf(zp - z));            // soft value
    if (hard)                                            // straight-through fwd
      y = (s == 0) ? ((z >= zp) ? 1.0f : 0.0f) : ((z > zp) ? 1.0f : 0.0f);
    all_lag[((size_t)b * KORD + k) * NOUT + cc] = y;
    if (s == 0) coeffs[((size_t)b * KORD + k) * (NVAR * NVAR) + (cc >> 1)] = y;
  }
}

// ----------------------------- BatchNorm -----------------------------------
__global__ void bn_stats_kernel(const float* __restrict__ h1,
                                const float* __restrict__ gamma,
                                const float* __restrict__ beta,
                                float* __restrict__ scale,
                                float* __restrict__ shift) {
  __shared__ float ssum[256];
  __shared__ float ssq[256];
  const int j = blockIdx.x;
  const int t = threadIdx.x;
  float s = 0.0f, q = 0.0f;
  for (int b = t; b < BSZ; b += 256) {
    const float v = h1[(size_t)b * HID + j];
    s += v; q += v * v;
  }
  ssum[t] = s; ssq[t] = q;
  __syncthreads();
  for (int off = 128; off > 0; off >>= 1) {
    if (t < off) { ssum[t] += ssum[t + off]; ssq[t] += ssq[t + off]; }
    __syncthreads();
  }
  if (t == 0) {
    const float mean = ssum[0] * (1.0f / BSZ);
    const float var  = ssq[0] * (1.0f / BSZ) - mean * mean;   // biased, as torch BN
    const float sc   = gamma[j] * rsqrtf(var + BN_EPS);
    scale[j] = sc;
    shift[j] = beta[j] - mean * sc;
  }
}

__global__ void bn_apply_kernel(const float* __restrict__ h1,
                                const float* __restrict__ scale,
                                const float* __restrict__ shift,
                                _Float16* __restrict__ out) {
  const int tid = blockIdx.x * blockDim.x + threadIdx.x;
  const int j = tid & (HID - 1);
  out[tid] = (_Float16)(h1[tid] * scale[j] + shift[j]);
}

// ------------------------ per-variable lag MLPs ----------------------------
// One wave per (b,v): lane o computes hidden units o and o+32 of the 64-wide
// ReLU layer, then cross-lane reduce for the scalar prediction.
__global__ void lag_mlp_kernel(const float* __restrict__ inputs,
                               const float* __restrict__ coeffs,
                               const float* __restrict__ src_pred,
                               const float* __restrict__ tgt_pred,
                               const int*   __restrict__ is_source,
                               const float* __restrict__ LW1,
                               const float* __restrict__ Lb1,
                               const float* __restrict__ LW2,
                               const float* __restrict__ Lb2,
                               float* __restrict__ preds, int k) {
  const int lane = threadIdx.x & 31;
  const int wid  = blockIdx.x * (blockDim.x >> 5) + (threadIdx.x >> 5);
  const int b = wid >> 5;
  const int v = wid & 31;
  const float* st = coeffs + ((size_t)b * KORD + k) * (NVAR * NVAR) + v * NVAR;
  const float* in = inputs + ((size_t)b * KORD + k) * NVAR;
  const float* dp = ((*is_source) ? src_pred : tgt_pred) + k * DDIM;
  const float* w1 = LW1 + (size_t)(k * NVAR + v) * (NVAR + DDIM) * 64;
  float a0 = 0.0f, a1 = 0.0f;
#pragma unroll
  for (int i = 0; i < NVAR + DDIM; ++i) {
    const float val = (i < NVAR) ? st[i] * in[i] : dp[i - NVAR];
    const float* w = w1 + i * 64;
    a0 += val * w[lane];
    a1 += val * w[lane + 32];
  }
  const float* lb1 = Lb1 + (size_t)(k * NVAR + v) * 64;
  const float* lw2 = LW2 + (size_t)(k * NVAR + v) * 64;
  const float h0 = fmaxf(a0 + lb1[lane], 0.0f);
  const float h1 = fmaxf(a1 + lb1[lane + 32], 0.0f);
  float lp = h0 * lw2[lane] + h1 * lw2[lane + 32];
  for (int off = 16; off > 0; off >>= 1) lp += __shfl_down(lp, off, 32);
  if (lane == 0) preds[(size_t)b * NVAR + v] += lp + Lb2[k * NVAR + v];
}

__global__ void zero_kernel(float* __restrict__ p, int n) {
  const int t = blockIdx.x * blockDim.x + threadIdx.x;
  if (t < n) p[t] = 0.0f;
}

// ------------------------------- launcher ----------------------------------
extern "C" void kernel_launch(void* const* d_in, const int* in_sizes, int n_in,
                              void* d_out, int out_size, void* d_ws, size_t ws_size,
                              hipStream_t stream) {
  (void)in_sizes; (void)n_in; (void)out_size; (void)ws_size;

  const float* inputs   = (const float*)d_in[0];
  const float* gumbel_u = (const float*)d_in[1];
  const float* src_var  = (const float*)d_in[2];
  const float* tgt_var  = (const float*)d_in[3];
  const float* src_pred = (const float*)d_in[4];
  const float* tgt_pred = (const float*)d_in[5];
  const float* W0 = (const float*)d_in[6];
  const float* b0 = (const float*)d_in[7];
  const float* W1 = (const float*)d_in[8];
  const float* b1 = (const float*)d_in[9];
  const float* bn_gamma = (const float*)d_in[10];
  const float* bn_beta  = (const float*)d_in[11];
  const float* W2 = (const float*)d_in[12];
  const float* b2 = (const float*)d_in[13];
  const float* LW1 = (const float*)d_in[14];
  const float* Lb1 = (const float*)d_in[15];
  const float* LW2 = (const float*)d_in[16];
  const float* Lb2 = (const float*)d_in[17];
  const int* hard      = (const int*)d_in[18];
  const int* is_source = (const int*)d_in[19];

  float* preds   = (float*)d_out;                                  // [B,NV]
  float* coeffs  = preds  + (size_t)BSZ * NVAR;                    // [B,5,32,32]
  float* all_lag = coeffs + (size_t)BSZ * KORD * NVAR * NVAR;      // [B,5,32,32,2]

  char* ws = (char*)d_ws;                     // ~11.3 MB used
  _Float16* x_f16  = (_Float16*)(ws + 0x000000);  // B*192 f16  (1.5 MB)
  _Float16* h0_f16 = (_Float16*)(ws + 0x180000);  // B*256 f16  (2 MB)
  float*    h1_f32 = (float*)   (ws + 0x380000);  // B*256 f32  (4 MB)
  _Float16* h1n    = (_Float16*)(ws + 0x780000);  // B*256 f16  (2 MB)
  _Float16* W0p    = (_Float16*)(ws + 0x980000);  // 192*256 f16
  _Float16* W1p    = (_Float16*)(ws + 0x9A0000);  // 256*256 f16
  _Float16* W2p    = (_Float16*)(ws + 0x9C0000);  // 256*2048 f16
  float*    scale  = (float*)   (ws + 0xAC0000);  // 256 f32
  float*    shift  = (float*)   (ws + 0xAC0400);  // 256 f32

  zero_kernel<<<(BSZ * NVAR) / 256, 256, 0, stream>>>(preds, BSZ * NVAR);

  for (int k = 0; k < KORD; ++k) {
    pack_w_kernel<<<(KP0 * HID) / 256, 256, 0, stream>>>(
        W0 + (size_t)k * INDIM * HID, W0p, INDIM, KP0, HID);
    pack_w_kernel<<<(HID * HID) / 256, 256, 0, stream>>>(
        W1 + (size_t)k * HID * HID, W1p, HID, HID, HID);
    pack_w_kernel<<<(HID * NOUT) / 256, 256, 0, stream>>>(
        W2 + (size_t)k * HID * NOUT, W2p, HID, HID, NOUT);

    build_x_kernel<<<BSZ, KP0, 0, stream>>>(inputs, coeffs, src_var, tgt_var,
                                            is_source, x_f16, k);

    dim3 g01(BSZ / 16, HID / 128);
    gemm_wmma_kernel<KP0><<<g01, 256, 0, stream>>>(x_f16, W0p, b0 + k * HID,
                                                   nullptr, h0_f16, HID);
    gemm_wmma_kernel<HID><<<g01, 256, 0, stream>>>(h0_f16, W1p, b1 + k * HID,
                                                   h1_f32, nullptr, HID);

    bn_stats_kernel<<<HID, 256, 0, stream>>>(h1_f32, bn_gamma + k * HID,
                                             bn_beta + k * HID, scale, shift);
    bn_apply_kernel<<<(BSZ * HID) / 256, 256, 0, stream>>>(h1_f32, scale, shift, h1n);

    dim3 g2(BSZ / 16, NOUT / 128);
    gemm2_gumbel_kernel<<<g2, 256, 0, stream>>>(
        h1n, W2p, b2 + k * NOUT, gumbel_u + (size_t)k * BSZ * NOUT, hard,
        coeffs, all_lag, k);

    lag_mlp_kernel<<<(BSZ * NVAR) / 8, 256, 0, stream>>>(
        inputs, coeffs, src_pred, tgt_pred, is_source,
        LW1, Lb1, LW2, Lb2, preds, k);
  }
}
